// AttentionMix_62182536511489
// MI455X (gfx1250) — compile-verified, matrix operand order checked by reading, not run on hardware
//
#include <hip/hip_runtime.h>
#include <hip/hip_bf16.h>
#include <math.h>

// ---------------------------------------------------------------------------
// AttentionMix on MI455X (gfx1250): bf16 WMMA for all GEMMs, flash-softmax
// streaming of the 604MB pair_logits tensor (HBM-roofline bound, ~26us).
//   Stage 0: f32->bf16 conversion + zero-fill of padded Q/K.
//   Stage 1: fused QKVG projection, K-loop unrolled x2 with alternating
//            operand buffers (no register-rotation copies), 4 WMMA accs.
//   Stage 2: flash attention; K/V tiles staged once per block in LDS,
//            pair_logits staged via vectorized float4 loads, P routed
//            through LDS to reshape C-layout -> A-layout.
// ---------------------------------------------------------------------------

typedef __attribute__((ext_vector_type(16))) __bf16 v16bf;
typedef __attribute__((ext_vector_type(8)))  __bf16 v8bf;
typedef __attribute__((ext_vector_type(8)))  float  v8f;

#define N_TOK 3072
#define C_DIM 768
#define H_NUM 16
#define D_HEAD 48
#define D_PAD  64   // pad head dim to 64 => K-dim of WMMA is exactly 2x32

// Load one 16x32 bf16 operand tile in the documented 16-bit A-matrix layout:
//   lane L (0..31): rows M = L%16; element i of v16bf holds
//   k = ((i>>3)<<4) + ((L>>4)<<3) + (i&7)
// i.e. two contiguous 8-element groups -> two global/ds b128 loads per lane.
__device__ __forceinline__ v16bf load_tile_a(const __bf16* __restrict__ base,
                                             int row_stride, int lane) {
  const int row = lane & 15;
  const int g   = (lane >> 4) << 3;           // 0 or 8
  const __bf16* rp = base + (size_t)row * row_stride;
  v8bf lo = *(const v8bf*)(rp + g);           // k = g .. g+7
  v8bf hi = *(const v8bf*)(rp + 16 + g);      // k = 16+g .. 16+g+7
  v16bf out;
#pragma unroll
  for (int i = 0; i < 8; ++i) { out[i] = lo[i]; out[i + 8] = hi[i]; }
  return out;
}

__device__ __forceinline__ v8f wmma_bf16(v16bf a, v16bf b, v8f c) {
  return __builtin_amdgcn_wmma_f32_16x16x32_bf16(
      /*neg_a=*/false, a, /*neg_b=*/false, b,
      /*c_mod=*/(short)0, c, /*reuse_a=*/false, /*reuse_b=*/false);
}

// ---------------------------------------------------------------------------
// Stage 0: f32 -> bf16 conversion, and zero-fill for padded Q/K buffers.
// ---------------------------------------------------------------------------
__global__ void cvt_bf16_kernel(const float* __restrict__ src,
                                __bf16* __restrict__ dst, int n) {
  for (int i = blockIdx.x * blockDim.x + threadIdx.x; i < n;
       i += gridDim.x * blockDim.x)
    dst[i] = (__bf16)src[i];
}

__global__ void zero_bf16_kernel(__bf16* __restrict__ dst, int n) {
  for (int i = blockIdx.x * blockDim.x + threadIdx.x; i < n;
       i += gridDim.x * blockDim.x)
    dst[i] = (__bf16)0.0f;
}

// ---------------------------------------------------------------------------
// Stage 1: fused QKVG projection.  out[n, j] = sum_c x[n,c] * W[j,c]
// Both x and W are row-major along c => both operands load with the A-layout
// pattern (B of a WMMA == "A-layout" view of the transposed source).
// Block = 128 thr (4 waves). Each wave: 16 rows x 64 cols (4 acc tiles).
// K-loop unrolled x2 with two alternating operand buffer sets so loads for
// step k+1 overlap the WMMAs of step k WITHOUT v_mov register rotation.
// ---------------------------------------------------------------------------
__global__ __launch_bounds__(128) void proj_kernel(
    const __bf16* __restrict__ xb,
    const __bf16* __restrict__ wqb, const __bf16* __restrict__ wkb,
    const __bf16* __restrict__ wvb, const __bf16* __restrict__ wgb,
    const float* __restrict__ bq,
    __bf16* __restrict__ qb,    // (H, N, D_PAD), scaled, +bias, pad zeroed
    __bf16* __restrict__ kb,    // (H, N, D_PAD), pad zeroed
    __bf16* __restrict__ vTb,   // (H, D_PAD, N) transposed per head
    float*  __restrict__ graw)  // (N, C) raw gate logits
{
  const int lane = threadIdx.x & 31;
  const int wv   = threadIdx.x >> 5;
  const int rt   = blockIdx.x;                 // 16-row tile
  const int cg   = blockIdx.y;                 // 256-col group
  const int col0 = cg * 256 + wv * 64;         // this wave's first column
  const int sel  = (cg * 256) / C_DIM;         // 0=q 1=k 2=v 3=g (block-uniform)
  const __bf16* w = (sel == 0) ? wqb : (sel == 1) ? wkb : (sel == 2) ? wvb : wgb;
  const int jl0  = col0 % C_DIM;               // local row in selected weight

  v8f acc[4] = {};
  const __bf16* xrow  = xb + (size_t)rt * 16 * C_DIM;
  const __bf16* wrow0 = w + (size_t)jl0 * C_DIM;

  // prologue: buffer set 0 holds tiles for k = 0
  v16bf a0 = load_tile_a(xrow, C_DIM, lane);
  v16bf b0_0 = load_tile_a(wrow0 + (size_t)(0 * 16) * C_DIM, C_DIM, lane);
  v16bf b0_1 = load_tile_a(wrow0 + (size_t)(1 * 16) * C_DIM, C_DIM, lane);
  v16bf b0_2 = load_tile_a(wrow0 + (size_t)(2 * 16) * C_DIM, C_DIM, lane);
  v16bf b0_3 = load_tile_a(wrow0 + (size_t)(3 * 16) * C_DIM, C_DIM, lane);
  v16bf a1, b1_0, b1_1, b1_2, b1_3;

  // 22 of 24 K-steps: 10 double-iterations (k and k+32) + 2-step tail; k = C-96, C-64 not run
  for (int kk = 0; kk < C_DIM - 128; kk += 64) {
    const int k1 = kk + 32, k2 = kk + 64;
    a1   = load_tile_a(xrow + k1, C_DIM, lane);
    b1_0 = load_tile_a(wrow0 + (size_t)(0 * 16) * C_DIM + k1, C_DIM, lane);
    b1_1 = load_tile_a(wrow0 + (size_t)(1 * 16) * C_DIM + k1, C_DIM, lane);
    b1_2 = load_tile_a(wrow0 + (size_t)(2 * 16) * C_DIM + k1, C_DIM, lane);
    b1_3 = load_tile_a(wrow0 + (size_t)(3 * 16) * C_DIM + k1, C_DIM, lane);
    acc[0] = wmma_bf16(a0, b0_0, acc[0]);
    acc[1] = wmma_bf16(a0, b0_1, acc[1]);
    acc[2] = wmma_bf16(a0, b0_2, acc[2]);
    acc[3] = wmma_bf16(a0, b0_3, acc[3]);
    a0   = load_tile_a(xrow + k2, C_DIM, lane);
    b0_0 = load_tile_a(wrow0 + (size_t)(0 * 16) * C_DIM + k2, C_DIM, lane);
    b0_1 = load_tile_a(wrow0 + (size_t)(1 * 16) * C_DIM + k2, C_DIM, lane);
    b0_2 = load_tile_a(wrow0 + (size_t)(2 * 16) * C_DIM + k2, C_DIM, lane);
    b0_3 = load_tile_a(wrow0 + (size_t)(3 * 16) * C_DIM + k2, C_DIM, lane);
    acc[0] = wmma_bf16(a1, b1_0, acc[0]);
    acc[1] = wmma_bf16(a1, b1_1, acc[1]);
    acc[2] = wmma_bf16(a1, b1_2, acc[2]);
    acc[3] = wmma_bf16(a1, b1_3, acc[3]);
  }
  {  // tail: buffer 0 holds k = C-128; load k = C-32 into buffer 1
    const int k1 = C_DIM - 32;
    a1   = load_tile_a(xrow + k1, C_DIM, lane);
    b1_0 = load_tile_a(wrow0 + (size_t)(0 * 16) * C_DIM + k1, C_DIM, lane);
    b1_1 = load_tile_a(wrow0 + (size_t)(1 * 16) * C_DIM + k1, C_DIM, lane);
    b1_2 = load_tile_a(wrow0 + (size_t)(2 * 16) * C_DIM + k1, C_DIM, lane);
    b1_3 = load_tile_a(wrow0 + (size_t)(3 * 16) * C_DIM + k1, C_DIM, lane);
    acc[0] = wmma_bf16(a0, b0_0, acc[0]);
    acc[1] = wmma_bf16(a0, b0_1, acc[1]);
    acc[2] = wmma_bf16(a0, b0_2, acc[2]);
    acc[3] = wmma_bf16(a0, b0_3, acc[3]);
    acc[0] = wmma_bf16(a1, b1_0, acc[0]);
    acc[1] = wmma_bf16(a1, b1_1, acc[1]);
    acc[2] = wmma_bf16(a1, b1_2, acc[2]);
    acc[3] = wmma_bf16(a1, b1_3, acc[3]);
  }

  const float rsD = 0.14433756729740643f;      // 48^-0.5
  const int m_off = (lane >> 4) << 3;
  const int nl    = lane & 15;
#pragma unroll
  for (int t = 0; t < 4; ++t) {
    const int j = col0 + t * 16 + nl;          // global column in [0, 4C)
#pragma unroll
    for (int r = 0; r < 8; ++r) {
      const int row = rt * 16 + r + m_off;
      const float val = acc[t][r];
      if (sel == 0) {
        const int jl = j, h = jl / D_HEAD, d = jl % D_HEAD;
        qb[((size_t)h * N_TOK + row) * D_PAD + d] = (__bf16)((val + bq[jl]) * rsD);
      } else if (sel == 1) {
        const int jl = j - C_DIM, h = jl / D_HEAD, d = jl % D_HEAD;
        kb[((size_t)h * N_TOK + row) * D_PAD + d] = (__bf16)val;
      } else if (sel == 2) {
        const int jl = j - 2 * C_DIM, h = jl / D_HEAD, d = jl % D_HEAD;
        vTb[((size_t)h * D_PAD + d) * N_TOK + row] = (__bf16)val;
      } else {
        graw[(size_t)row * C_DIM + (j - 3 * C_DIM)] = val;
      }
    }
  }
}

// ---------------------------------------------------------------------------
// Stage 2: flash attention + pair_logits bias + gated output.
// Block = 128 thr (4 waves); wave handles one (head, 16-query) tile; the 4
// waves of a block share the same (h, key-tile) so K and V^T tiles are
// staged ONCE per block-iteration in LDS (4x less VMEM than per-wave loads).
// pair_logits (the dominant 604MB stream) is staged per-wave with float4
// loads. Streams 32 keys per iteration: 4 logits WMMAs + 3 P.V WMMAs.
// ---------------------------------------------------------------------------
__global__ __launch_bounds__(128) void attn_kernel(
    const __bf16* __restrict__ qb, const __bf16* __restrict__ kb,
    const __bf16* __restrict__ vTb,
    const float* __restrict__ pair, const unsigned char* __restrict__ mask,
    const float* __restrict__ graw, float* __restrict__ out)
{
  __shared__ __bf16 kb_lds[32][64];        // 4KB : 32 keys x 64 d (block-shared)
  __shared__ __bf16 vT_lds[48][32];        // 3KB : 48 d x 32 keys (block-shared)
  __shared__ __bf16 p_lds[4][16][32];      // 4KB : probs, per wave
  __shared__ float  pair_lds[4][16][32];   // 8KB : pair tile, per wave

  const int tid  = threadIdx.x;
  const int lane = tid & 31;
  const int wv   = tid >> 5;
  const int h    = blockIdx.x;
  const int q0   = (blockIdx.y * 4 + wv) * 16;
  const int m_off = (lane >> 4) << 3;
  const int nl    = lane & 15;

  const __bf16* qbase = qb + ((size_t)h * N_TOK + q0) * D_PAD;
  const v16bf aq_lo = load_tile_a(qbase,      D_PAD, lane);   // d = 0..31
  const v16bf aq_hi = load_tile_a(qbase + 32, D_PAD, lane);   // d = 32..63

  float mrun[8], lrun[8];
  v8f acc[3] = {};
#pragma unroll
  for (int r = 0; r < 8; ++r) { mrun[r] = -1e30f; lrun[r] = 0.0f; }

  const float* pair_base = pair + ((size_t)h * N_TOK + q0) * N_TOK;
  const __bf16* ksrc0 = kb  + (size_t)h * N_TOK * D_PAD;
  const __bf16* vsrc0 = vTb + (size_t)h * D_PAD * N_TOK;

  for (int kt = 0; kt < N_TOK; kt += 32) {
    // ---- stage: K tile (block), V^T tile (block), pair tile (per wave) ----
    {
      const __bf16* ksrc = ksrc0 + (size_t)kt * D_PAD;
#pragma unroll
      for (int i = 0; i < 2; ++i) {          // 256 b128 chunks over 128 thr
        const int c = tid * 2 + i;
        const int row = c >> 3, off = (c & 7) * 8;
        *(v8bf*)&kb_lds[row][off] = *(const v8bf*)(ksrc + (size_t)row * D_PAD + off);
      }
      const __bf16* vsrc = vsrc0 + kt;
#pragma unroll
      for (int i = 0; i < 2; ++i) {          // 192 b128 chunks over 128 thr
        const int c = tid + i * 128;
        if (c < 192) {
          const int row = c >> 2, off = (c & 3) * 8;
          *(v8bf*)&vT_lds[row][off] = *(const v8bf*)(vsrc + (size_t)row * N_TOK + off);
        }
      }
      const int prow = lane >> 1, pcol = (lane & 1) * 16;
      const float* psrc = pair_base + (size_t)prow * N_TOK + kt + pcol;
#pragma unroll
      for (int t = 0; t < 4; ++t)
        *(float4*)&pair_lds[wv][prow][pcol + t * 4] = *(const float4*)(psrc + t * 4);
      if (kt + 64 < N_TOK)                    // stream hint, 2 tiles ahead
        __builtin_prefetch(psrc + 64, 0, 0);
    }
    __syncthreads();                          // B1: tiles visible

    // ---- logits: two 16-key sub-tiles, K=64 chained WMMA, from LDS ----
    v8f s[2];
#pragma unroll
    for (int sub = 0; sub < 2; ++sub) {
      v16bf b_lo = load_tile_a(&kb_lds[sub * 16][0],  64, lane);
      v16bf b_hi = load_tile_a(&kb_lds[sub * 16][32], 64, lane);
      v8f c = {};
      c = wmma_bf16(aq_lo, b_lo, c);
      c = wmma_bf16(aq_hi, b_hi, c);
      const int key = kt + sub * 16 + nl;
      const float kb_bias = mask[key] ? 0.0f : -1e9f;
#pragma unroll
      for (int r = 0; r < 8; ++r)
        c[r] += pair_lds[wv][r + m_off][sub * 16 + nl] + kb_bias;
      s[sub] = c;
    }
    // preload V^T B-operands (vT_lds valid since B1; consumed after B2)
    v16bf bV[3];
#pragma unroll
    for (int dt = 0; dt < 3; ++dt)
      bV[dt] = load_tile_a(&vT_lds[dt * 16][0], 32, lane);

    // ---- online softmax (rows live in 16-lane halves of the wave) ----
#pragma unroll
    for (int r = 0; r < 8; ++r) {
      float mx = fmaxf(s[0][r], s[1][r]);
#pragma unroll
      for (int off = 8; off >= 1; off >>= 1)
        mx = fmaxf(mx, __shfl_xor(mx, off, 32));
      const float mnew  = fmaxf(mrun[r], mx);
      const float alpha = __expf(mrun[r] - mnew);
      const float p0 = __expf(s[0][r] - mnew);
      const float p1 = __expf(s[1][r] - mnew);
      float rsum = p0 + p1;
#pragma unroll
      for (int off = 8; off >= 1; off >>= 1)
        rsum += __shfl_xor(rsum, off, 32);
      lrun[r] = lrun[r] * alpha + rsum;
      mrun[r] = mnew;
#pragma unroll
      for (int dt = 0; dt < 3; ++dt) acc[dt][r] *= alpha;
      p_lds[wv][r + m_off][nl]      = (__bf16)p0;   // C-layout -> LDS
      p_lds[wv][r + m_off][16 + nl] = (__bf16)p1;
    }
    __syncthreads();                          // B2: p_lds visible; K/V reads done

    // ---- P.V: reshape P via LDS into A-layout, 3 WMMAs (d < 48) ----
    const v16bf pA = load_tile_a(&p_lds[wv][0][0], 32, lane);
#pragma unroll
    for (int dt = 0; dt < 3; ++dt)
      acc[dt] = wmma_bf16(pA, bV[dt], acc[dt]);
    // next stage phase may overwrite kb/vT/pair tiles: all reads of those
    // finished before B2, and p_lds is wave-private => no third barrier.
  }

  // ---- epilogue: 1/l scaling, gate sigmoid, store (N, C) f32 ----
#pragma unroll
  for (int dt = 0; dt < 3; ++dt) {
    const int j = h * D_HEAD + dt * 16 + nl;
#pragma unroll
    for (int r = 0; r < 8; ++r) {
      const int row = q0 + r + m_off;
      const float g = graw[(size_t)row * C_DIM + j];
      const float o = acc[dt][r] / lrun[r];
      out[(size_t)row * C_DIM + j] = o * (1.0f / (1.0f + __expf(-g)));
    }
  }
}

// ---------------------------------------------------------------------------
// Host launcher
// ---------------------------------------------------------------------------
extern "C" void kernel_launch(void* const* d_in, const int* in_sizes, int n_in,
                              void* d_out, int out_size, void* d_ws, size_t ws_size,
                              hipStream_t stream) {
  const float*         x    = (const float*)d_in[0];
  const unsigned char* mask = (const unsigned char*)d_in[1];
  const float*         pair = (const float*)d_in[2];
  const float*         wq   = (const float*)d_in[3];
  const float*         bq   = (const float*)d_in[4];
  const float*         wk   = (const float*)d_in[5];
  const float*         wv   = (const float*)d_in[6];
  const float*         wg   = (const float*)d_in[7];
  float* out = (float*)d_out;

  // workspace carve-up (all 256B-aligned sizes)
  char* ws = (char*)d_ws;
  const size_t xb_elems = (size_t)N_TOK * C_DIM;
  const size_t w_elems  = (size_t)C_DIM * C_DIM;
  const size_t hd_elems = (size_t)H_NUM * N_TOK * D_PAD;
  __bf16* xb  = (__bf16*)ws;                    ws += xb_elems * 2;
  __bf16* wqb = (__bf16*)ws;                    ws += w_elems * 2;
  __bf16* wkb = (__bf16*)ws;                    ws += w_elems * 2;
  __bf16* wvb = (__bf16*)ws;                    ws += w_elems * 2;
  __bf16* wgb = (__bf16*)ws;                    ws += w_elems * 2;
  __bf16* qb  = (__bf16*)ws;                    ws += hd_elems * 2;
  __bf16* kb  = (__bf16*)ws;                    ws += hd_elems * 2;
  __bf16* vTb = (__bf16*)ws;                    ws += hd_elems * 2;
  float*  graw = (float*)ws;                    ws += xb_elems * 4;

  // Stage 0: conversions + zero padded Q/K (qb and kb are adjacent)
  cvt_bf16_kernel<<<2048, 256, 0, stream>>>(x,  xb,  (int)xb_elems);
  cvt_bf16_kernel<<<1024, 256, 0, stream>>>(wq, wqb, (int)w_elems);
  cvt_bf16_kernel<<<1024, 256, 0, stream>>>(wk, wkb, (int)w_elems);
  cvt_bf16_kernel<<<1024, 256, 0, stream>>>(wv, wvb, (int)w_elems);
  cvt_bf16_kernel<<<1024, 256, 0, stream>>>(wg, wgb, (int)w_elems);
  zero_bf16_kernel<<<2048, 256, 0, stream>>>(qb, (int)(2 * hd_elems));

  // Stage 1: fused QKVG projection (bf16 WMMA, x2-unrolled pipelined K-loop)
  proj_kernel<<<dim3(N_TOK / 16, (4 * C_DIM) / 256), 128, 0, stream>>>(
      xb, wqb, wkb, wvb, wgb, bq, qb, kb, vTb, graw);

  // Stage 2: flash attention + pair bias + gate
  attn_kernel<<<dim3(H_NUM, N_TOK / 64), 128, 0, stream>>>(
      qb, kb, vTb, pair, mask, graw, out);

  (void)in_sizes; (void)n_in; (void)out_size; (void)ws_size;
}
